// CausalSelfAttention_63282048139683
// MI455X (gfx1250) — matrix operand order repaired and checked by reading, not compiled, not run on hardware
//
#include <hip/hip_runtime.h>
#include <stdint.h>

#define NUM_HEADS 16
#define HEAD_DIM  64
#define BATCH     2
#define SEQ       2048
#define EMBED     1024
#define BH        (BATCH * NUM_HEADS)     // 32
#define MROWS     (BATCH * SEQ)           // 4096

typedef __attribute__((ext_vector_type(16))) __bf16 bf16x16;
typedef __attribute__((ext_vector_type(8)))  float  f32x8;

union FragBF { bf16x16 v; unsigned short u[16]; uint4 q[2]; };

__device__ __forceinline__ unsigned short f2bf(float f) {
  union { float f; unsigned int u; } v; v.f = f;
  return (unsigned short)((v.u + 0x7FFFu + ((v.u >> 16) & 1u)) >> 16);
}

// Async 16B global -> LDS copy (no VGPR round-trip, tracked by ASYNCcnt).
// LDS byte address = low 32 bits of the generic pointer (aperture truncation).
__device__ __forceinline__ void async_copy16(void* lds_dst, const void* gsrc) {
  asm volatile("global_load_async_to_lds_b128 %0, %1, off"
               :: "v"((unsigned int)(uintptr_t)lds_dst), "v"(gsrc)
               : "memory");
}
__device__ __forceinline__ void async_wait0() {
  asm volatile("s_wait_asynccnt 0" ::: "memory");
}

// ---------------------------------------------------------------------------
// fp32 -> bf16 elementwise convert
// ---------------------------------------------------------------------------
__global__ void k_f32_to_bf16(const float* __restrict__ in,
                              unsigned short* __restrict__ out, int n) {
  int i = blockIdx.x * blockDim.x + threadIdx.x;
  if (i < n) out[i] = f2bf(in[i]);
}

// fp32 [K,N] -> bf16 transposed [N,K] (output-coalesced)
__global__ void k_f32_to_bf16_t(const float* __restrict__ in,
                                unsigned short* __restrict__ out,
                                int K, int N) {
  int i = blockIdx.x * blockDim.x + threadIdx.x;
  if (i < K * N) {
    int n = i / K, k = i - n * K;
    out[i] = f2bf(in[(size_t)k * N + n]);
  }
}

// ---------------------------------------------------------------------------
// C[M,N] f32 = A[M,K] bf16 * Bt[N,K] bf16 (B pre-transposed).
// M%128==0, N%128==0, K%32==0.  256 threads = 8 waves; 128x128 C tile.
// Each wave owns a 2x4 grid of 16x16 subtiles: 8 WMMAs per 32-k step from
// 2 A-frags + 4 B-frags.  Tiles staged into LDS in WMMA fragment order via
// async global->LDS copies; lanes read 32B fragments contiguously.
// ---------------------------------------------------------------------------
__global__ __launch_bounds__(256) void k_gemm_bf16(
    const unsigned short* __restrict__ A, const unsigned short* __restrict__ Bt,
    float* __restrict__ C, int Mdim, int Ndim, int Kdim) {
  __shared__ unsigned short aF[8][32][16];   // [row-block][lane][elem]
  __shared__ unsigned short bF[8][32][16];   // [col-block][lane][elem]
  const int tid  = threadIdx.x;
  const int lane = tid & 31, wave = tid >> 5;
  const int hi   = lane >> 4, ln = lane & 15;
  const int m0 = blockIdx.y * 128, n0 = blockIdx.x * 128;
  const int r0 = (wave & 3) * 2;    // row blocks r0, r0+1
  const int c0 = (wave >> 2) * 4;   // col blocks c0 .. c0+3

  f32x8 acc[2][4];
  #pragma unroll
  for (int i = 0; i < 2; ++i)
    #pragma unroll
    for (int j = 0; j < 4; ++j) acc[i][j] = (f32x8){};

  // Staging: thread -> rows sr, sr+64; k-chunk sc in {0,8,16,24}; 16B each.
  const int sr = tid >> 2;
  const int sc = (tid & 3) * 8;
  // A fragment layout: lane = (k/8 & 1)*16 + row%16 ; elem = k%8 + 8*(k/16)
  const int aln = ((sc >> 3) & 1) * 16 + (sr & 15);
  const int ael = (sc & 16) ? 8 : 0;
  unsigned short* adst0 = &aF[(sr >> 4)][aln][ael];
  unsigned short* adst1 = &aF[(sr >> 4) + 4][aln][ael];
  // B fragment layout: lane = (k/8 & 1)*16 + col%16 ; elem = k%8 + 8*(k/16)
  const int bln = ((sc >> 3) & 1) * 16 + (sr & 15);
  const int bel = (sc & 16) ? 8 : 0;
  unsigned short* bdst0 = &bF[(sr >> 4)][bln][bel];
  unsigned short* bdst1 = &bF[(sr >> 4) + 4][bln][bel];

  const unsigned short* ag = A  + (size_t)(m0 + sr) * Kdim + sc;
  const unsigned short* bg = Bt + (size_t)(n0 + sr) * Kdim + sc;
  const size_t half = (size_t)64 * Kdim;

  for (int k0 = 0; k0 < Kdim; k0 += 32) {
    async_copy16(adst0, ag + k0);
    async_copy16(adst1, ag + half + k0);
    async_copy16(bdst0, bg + k0);
    async_copy16(bdst1, bg + half + k0);
    if (k0 + 32 < Kdim) {
      __builtin_prefetch(ag + k0 + 32, 0, 1);
      __builtin_prefetch(ag + half + k0 + 32, 0, 1);
      __builtin_prefetch(bg + k0 + 32, 0, 1);
      __builtin_prefetch(bg + half + k0 + 32, 0, 1);
    }
    async_wait0();
    __syncthreads();

    bf16x16 av[2], bv[4];
    #pragma unroll
    for (int i = 0; i < 2; ++i) av[i] = *(const bf16x16*)(&aF[r0 + i][lane][0]);
    #pragma unroll
    for (int j = 0; j < 4; ++j) bv[j] = *(const bf16x16*)(&bF[c0 + j][lane][0]);

    #pragma unroll
    for (int i = 0; i < 2; ++i)
      #pragma unroll
      for (int j = 0; j < 4; ++j)
        acc[i][j] = __builtin_amdgcn_wmma_f32_16x16x32_bf16(
            false, av[i], false, bv[j], (short)0, acc[i][j], false, false);
    __syncthreads();
  }

  #pragma unroll
  for (int i = 0; i < 2; ++i) {
    #pragma unroll
    for (int j = 0; j < 4; ++j) {
      float* p = C + (size_t)(m0 + (r0 + i) * 16 + 8 * hi) * Ndim
                   + n0 + (c0 + j) * 16 + ln;
      #pragma unroll
      for (int e = 0; e < 8; ++e) { *p = acc[i][j][e]; p += Ndim; }
    }
  }
}

// ---------------------------------------------------------------------------
// RoPE on q,k + repack to bf16.  qkv f32 [b, s, 3*H*D].
// q,k -> [bh, s, d] bf16 ; v -> transposed [bh, d, s] bf16 (for PV B-frags).
// ---------------------------------------------------------------------------
__global__ void k_rope_pack(const float* __restrict__ qkv,
                            unsigned short* __restrict__ qb,
                            unsigned short* __restrict__ kb,
                            unsigned short* __restrict__ vt) {
  const int g = blockIdx.x * blockDim.x + threadIdx.x;   // BH*SEQ*32 exact
  const int pair = g & 31;
  const int s    = (g >> 5) & (SEQ - 1);
  const int bh   = g >> 16;
  const int b = bh >> 4, h = bh & 15;
  const int d0 = pair * 2;

  const size_t base = (size_t)(b * SEQ + s) * 3072 + h * 64 + d0;
  const float q0v = qkv[base + 0 * 1024], q1v = qkv[base + 0 * 1024 + 1];
  const float k0v = qkv[base + 1 * 1024], k1v = qkv[base + 1 * 1024 + 1];
  const float v0v = qkv[base + 2 * 1024], v1v = qkv[base + 2 * 1024 + 1];

  const float freq = expf(-(float)d0 * (9.210340371976184f / 64.0f));
  const float ang = (float)s * freq;
  float sn, cs;
  __sincosf(ang, &sn, &cs);

  const size_t o = ((size_t)bh * SEQ + s) * 64 + d0;
  qb[o]     = f2bf(q0v * cs - q1v * sn);
  qb[o + 1] = f2bf(q0v * sn + q1v * cs);
  kb[o]     = f2bf(k0v * cs - k1v * sn);
  kb[o + 1] = f2bf(k0v * sn + k1v * cs);
  vt[((size_t)bh * 64 + d0) * SEQ + s]     = f2bf(v0v);
  vt[((size_t)bh * 64 + d0 + 1) * SEQ + s] = f2bf(v1v);
}

// ---------------------------------------------------------------------------
// Flash attention: one wave per 16-row q tile, online softmax in registers.
// Causal mask, scale 1/8.  Output attn bf16 [b, s, h*d].
// ---------------------------------------------------------------------------
__global__ __launch_bounds__(256) void k_attn(
    const unsigned short* __restrict__ qb,
    const unsigned short* __restrict__ kb,
    const unsigned short* __restrict__ vt,
    unsigned short* __restrict__ attnb) {
  __shared__ unsigned short p_lds[8][16 * 32];   // per-wave P transpose buffer
  const int lane = threadIdx.x & 31, wave = threadIdx.x >> 5;
  const int hi = lane >> 4, ln = lane & 15;
  const int tile = blockIdx.x * 8 + wave;        // bh*128 + qtile
  const int bh = tile >> 7;
  const int q0 = (tile & 127) * 16;
  const int b = bh >> 4, h = bh & 15;
  const float scale = 0.125f;                    // 1/sqrt(64)

  // Q A-fragments (K = 0..31 and 32..63): 16B-aligned vector gathers
  FragBF aq0, aq1;
  {
    const uint4* q4 = (const uint4*)(qb + ((size_t)bh * SEQ + q0 + ln) * 64);
    aq0.q[0] = q4[2 * hi];     aq0.q[1] = q4[2 * hi + 1];
    aq1.q[0] = q4[4 + 2 * hi]; aq1.q[1] = q4[5 + 2 * hi];
  }

  f32x8 o0 = {}, o1 = {}, o2 = {}, o3 = {};
  float mrow[8], lrow[8];
  #pragma unroll
  for (int i = 0; i < 8; ++i) { mrow[i] = -3.0e38f; lrow[i] = 0.0f; }

  const int nch = (q0 + 15) / 32 + 1;            // kv chunks of 32
  unsigned short* pw = &p_lds[wave][0];
  const uint4* pr4 = (const uint4*)(pw + ln * 32);

  // Running pointers: advance by 32 kv rows per chunk.
  const unsigned short* kp = kb + (size_t)bh * SEQ * 64 + (size_t)ln * 64 + hi * 16;
  const unsigned short* vp = vt + (size_t)bh * 64 * SEQ + (size_t)ln * SEQ + hi * 16;

  for (int c = 0; c < nch; ++c) {
    const int kv0 = c * 32;
    const bf16x16 bk00 = *(const bf16x16*)(kp);
    const bf16x16 bk01 = *(const bf16x16*)(kp + 32);
    const bf16x16 bk10 = *(const bf16x16*)(kp + 1024);
    const bf16x16 bk11 = *(const bf16x16*)(kp + 1024 + 32);
    kp += 2048;

    f32x8 s0 = {}, s1 = {};
    s0 = __builtin_amdgcn_wmma_f32_16x16x32_bf16(false, aq0.v, false, bk00, (short)0, s0, false, false);
    s0 = __builtin_amdgcn_wmma_f32_16x16x32_bf16(false, aq1.v, false, bk01, (short)0, s0, false, false);
    s1 = __builtin_amdgcn_wmma_f32_16x16x32_bf16(false, aq0.v, false, bk10, (short)0, s1, false, false);
    s1 = __builtin_amdgcn_wmma_f32_16x16x32_bf16(false, aq1.v, false, bk11, (short)0, s1, false, false);

    float alpha[8];
    #pragma unroll
    for (int i = 0; i < 8; ++i) {
      const int row = q0 + i + 8 * hi;
      s0[i] = (kv0 + ln      <= row) ? s0[i] * scale : -1.0e9f;
      s1[i] = (kv0 + 16 + ln <= row) ? s1[i] * scale : -1.0e9f;
      float cm = fmaxf(s0[i], s1[i]);
      cm = fmaxf(cm, __shfl_xor(cm, 1, 32));
      cm = fmaxf(cm, __shfl_xor(cm, 2, 32));
      cm = fmaxf(cm, __shfl_xor(cm, 4, 32));
      cm = fmaxf(cm, __shfl_xor(cm, 8, 32));
      const float mn = fmaxf(mrow[i], cm);
      alpha[i] = __expf(mrow[i] - mn);
      const float p0 = __expf(s0[i] - mn);
      const float p1 = __expf(s1[i] - mn);
      float rs = p0 + p1;
      rs += __shfl_xor(rs, 1, 32);
      rs += __shfl_xor(rs, 2, 32);
      rs += __shfl_xor(rs, 4, 32);
      rs += __shfl_xor(rs, 8, 32);
      lrow[i] = lrow[i] * alpha[i] + rs;
      mrow[i] = mn;
      pw[(i + 8 * hi) * 32 + ln]      = f2bf(p0);
      pw[(i + 8 * hi) * 32 + 16 + ln] = f2bf(p1);
      o0[i] *= alpha[i]; o1[i] *= alpha[i]; o2[i] *= alpha[i]; o3[i] *= alpha[i];
    }
    asm volatile("s_wait_dscnt 0" ::: "memory");   // wave-local LDS transpose visible

    // P A-fragment (16 rows x 32 kv): two 16B LDS vector loads
    FragBF ap;
    ap.q[0] = pr4[2 * hi];
    ap.q[1] = pr4[2 * hi + 1];

    const bf16x16 bv0 = *(const bf16x16*)(vp + kv0);
    const bf16x16 bv1 = *(const bf16x16*)(vp + 16 * SEQ + kv0);
    const bf16x16 bv2 = *(const bf16x16*)(vp + 32 * SEQ + kv0);
    const bf16x16 bv3 = *(const bf16x16*)(vp + 48 * SEQ + kv0);
    o0 = __builtin_amdgcn_wmma_f32_16x16x32_bf16(false, ap.v, false, bv0, (short)0, o0, false, false);
    o1 = __builtin_amdgcn_wmma_f32_16x16x32_bf16(false, ap.v, false, bv1, (short)0, o1, false, false);
    o2 = __builtin_amdgcn_wmma_f32_16x16x32_bf16(false, ap.v, false, bv2, (short)0, o2, false, false);
    o3 = __builtin_amdgcn_wmma_f32_16x16x32_bf16(false, ap.v, false, bv3, (short)0, o3, false, false);
  }

  unsigned short* ob = attnb + (size_t)(b * SEQ + q0 + 8 * hi) * 1024 + h * 64 + ln;
  #pragma unroll
  for (int i = 0; i < 8; ++i) {
    const float inv = 1.0f / lrow[i];
    ob[0]  = f2bf(o0[i] * inv);
    ob[16] = f2bf(o1[i] * inv);
    ob[32] = f2bf(o2[i] * inv);
    ob[48] = f2bf(o3[i] * inv);
    ob += 1024;
  }
}

// ---------------------------------------------------------------------------
extern "C" void kernel_launch(void* const* d_in, const int* in_sizes, int n_in,
                              void* d_out, int out_size, void* d_ws, size_t ws_size,
                              hipStream_t stream) {
  (void)in_sizes; (void)n_in; (void)out_size; (void)ws_size;
  const float* x     = (const float*)d_in[0];
  const float* W_qkv = (const float*)d_in[1];
  const float* W_out = (const float*)d_in[2];
  float* out = (float*)d_out;

  char* ws = (char*)d_ws;
  size_t off = 0;
  auto take = [&](size_t bytes) -> char* {
    char* p = ws + off;
    off += (bytes + 255) & ~(size_t)255;
    return p;
  };
  unsigned short* xb    = (unsigned short*)take((size_t)MROWS * EMBED * 2);
  unsigned short* wb1t  = (unsigned short*)take((size_t)EMBED * 3072 * 2);   // [3072][1024]
  unsigned short* wb2t  = (unsigned short*)take((size_t)1024 * EMBED * 2);   // [1024][1024]
  float*          qkv   = (float*)take((size_t)MROWS * 3072 * 4);
  unsigned short* qb    = (unsigned short*)take((size_t)BH * SEQ * 64 * 2);
  unsigned short* kbf   = (unsigned short*)take((size_t)BH * SEQ * 64 * 2);
  unsigned short* vtb   = (unsigned short*)take((size_t)BH * SEQ * 64 * 2);
  unsigned short* attnb = (unsigned short*)take((size_t)MROWS * 1024 * 2);

  const int n1 = MROWS * EMBED;
  k_f32_to_bf16<<<(n1 + 255) / 256, 256, 0, stream>>>(x, xb, n1);
  const int n2 = EMBED * 3072;
  k_f32_to_bf16_t<<<(n2 + 255) / 256, 256, 0, stream>>>(W_qkv, wb1t, EMBED, 3072);
  const int n3 = EMBED * 1024;
  k_f32_to_bf16_t<<<(n3 + 255) / 256, 256, 0, stream>>>(W_out, wb2t, EMBED, 1024);

  dim3 g1(3072 / 128, MROWS / 128);
  k_gemm_bf16<<<g1, 256, 0, stream>>>(xb, wb1t, qkv, MROWS, 3072, EMBED);

  const int nr = BH * SEQ * 32;
  k_rope_pack<<<nr / 256, 256, 0, stream>>>(qkv, qb, kbf, vtb);

  k_attn<<<(BH * (SEQ / 16)) / 8, 256, 0, stream>>>(qb, kbf, vtb, attnb);

  dim3 g2(1024 / 128, MROWS / 128);
  k_gemm_bf16<<<g2, 256, 0, stream>>>(attnb, wb2t, out, MROWS, 1024, 1024);
}
